// DGMC_84275848282598
// MI455X (gfx1250) — matrix operand order repaired and checked by reading, not compiled, not run on hardware
//
#include <hip/hip_runtime.h>

// ---------------------------------------------------------------------------
// DGMC forward for MI455X (gfx1250, wave32, WMMA).
// Heavy GEMMs run through v_wmma_f32_16x16x32_f16 with f32 accumulation.
// Big GEMM is register-blocked: 32x64 output per wave (8 accumulators),
// 8 WMMAs per K-step with 2 A-frags x 4 B-frags reuse.
// ---------------------------------------------------------------------------

typedef __attribute__((ext_vector_type(16))) _Float16 v16h;
typedef __attribute__((ext_vector_type(8)))  _Float16 v8h;
typedef __attribute__((ext_vector_type(8)))  float    v8f;

namespace {
constexpr int kB    = 2048;
constexpr int kNUM  = 15;
constexpr int kN    = kB * kNUM;   // 30720 nodes per side
constexpr int kE    = kB * 30;     // 61440 edges per side
constexpr int kD    = 768;         // D_IN
constexpr int kR    = 32;          // R_IN == R_OUT
constexpr int kNC   = 2 * kD;      // fused [Wn1|Wr1] output cols = 1536
constexpr int kSteps = 10;
}

// ------------------------------- utilities --------------------------------

__global__ __launch_bounds__(256) void dgmc_zero_f32(float* p, long n) {
  long i = (long)blockIdx.x * blockDim.x + threadIdx.x;
  if (i < n) p[i] = 0.0f;
}

__global__ __launch_bounds__(256) void dgmc_zero_i32(int* p, long n) {
  long i = (long)blockIdx.x * blockDim.x + threadIdx.x;
  if (i < n) p[i] = 0;
}

__device__ __forceinline__ unsigned dgmc_hash(unsigned x) {
  x ^= x >> 16; x *= 0x7feb352dU;
  x ^= x >> 15; x *= 0x846ca68bU;
  x ^= x >> 16; return x;
}

// ------------------------- degree features (s/t) --------------------------

__global__ __launch_bounds__(256) void dgmc_deg_count(
    const int* __restrict__ ei, int* __restrict__ outd, int* __restrict__ ind) {
  int e = blockIdx.x * blockDim.x + threadIdx.x;
  if (e >= kE) return;
  atomicAdd(&outd[ei[e]], 1);        // ei[0][e] = src
  atomicAdd(&ind[ei[kE + e]], 1);    // ei[1][e] = dst
}

__global__ __launch_bounds__(256) void dgmc_deg_feats(
    const float* __restrict__ x, const float* __restrict__ embin,
    const float* __restrict__ embout, const int* __restrict__ ind,
    const int* __restrict__ outd, float* __restrict__ xdf,
    _Float16* __restrict__ xh) {
  long i = (long)blockIdx.x * blockDim.x + threadIdx.x;
  if (i >= (long)kN * kD) return;
  int n = (int)(i / kD), j = (int)(i % kD);
  int di = ind[n];  di = di > 63 ? 63 : di;
  int dd = outd[n]; dd = dd > 63 ? 63 : dd;
  float v = x[i] + embin[(size_t)di * kD + j] + embout[(size_t)dd * kD + j];
  xdf[i] = v;
  xh[i] = (_Float16)v;
}

// ----------------- fused weight convert+transpose to f16 ------------------
// Wct[n][k] = (n < 768 ? Wn1[k][n] : Wr1[k][n-768]), column-major weights so
// each WMMA B-fragment lane reads 16 contiguous halves (32B) along K.

__global__ __launch_bounds__(256) void dgmc_conv_w(
    const float* __restrict__ Wn1, const float* __restrict__ Wr1,
    _Float16* __restrict__ Wct) {
  long i = (long)blockIdx.x * blockDim.x + threadIdx.x;
  if (i >= (long)kNC * kD) return;
  int n = (int)(i / kD), k = (int)(i % kD);
  float v = (n < kD) ? Wn1[(size_t)k * kD + n] : Wr1[(size_t)k * kD + (n - kD)];
  Wct[(size_t)n * kD + k] = (_Float16)v;
}

// --------------------- big WMMA GEMM: C = A(f16) * Bt^T -------------------
// A: [kN][768] row-major f16, Bt: [1536][768] (transposed weights),
// C: [kN][1536] f32.
// Register-blocked: 8 waves/block as 4(M)x2(N); each wave owns a 32x64 tile
// (2 A-frags x 4 B-frags -> 8 f32 accumulators, 8 WMMAs per 32-wide K-step).

__global__ __launch_bounds__(256) void dgmc_gemm_wmma(
    const _Float16* __restrict__ A, const _Float16* __restrict__ Bt,
    float* __restrict__ C) {
  int wave = threadIdx.x >> 5;         // 0..7
  int lane = threadIdx.x & 31;
  int wm = wave & 3;                   // M sub-tile
  int wn = wave >> 2;                  // N sub-tile
  int m0 = blockIdx.x * 128 + wm * 32; // rows m0..m0+31 (two 16-row frags)
  int n0 = blockIdx.y * 128 + wn * 64; // cols n0..n0+63 (four 16-col frags)
  int l15 = lane & 15;
  int hi  = lane >> 4;
  // A fragment: row m = l15, halves j -> K = k0 + hi*16 + j
  const _Float16* pa0 = A + ((size_t)(m0 + l15)) * kD + hi * 16;
  const _Float16* pa1 = pa0 + (size_t)16 * kD;
  // B fragment: col n = l15, halves j -> K = k0 + hi*16 + j (contiguous 32B)
  const _Float16* pb0 = Bt + ((size_t)(n0 + l15)) * kD + hi * 16;

  v8f acc[8];
#pragma unroll
  for (int i = 0; i < 8; ++i)
#pragma unroll
    for (int q = 0; q < 8; ++q) acc[i][q] = 0.0f;

  for (int k0 = 0; k0 < kD; k0 += 32) {
    v16h av[2], bv[4];
    {
      v8h a00 = *(const v8h*)(pa0 + k0);
      v8h a01 = *(const v8h*)(pa0 + k0 + 8);
      v8h a10 = *(const v8h*)(pa1 + k0);
      v8h a11 = *(const v8h*)(pa1 + k0 + 8);
#pragma unroll
      for (int q = 0; q < 8; ++q) {
        av[0][q] = a00[q]; av[0][q + 8] = a01[q];
        av[1][q] = a10[q]; av[1][q + 8] = a11[q];
      }
    }
#pragma unroll
    for (int j = 0; j < 4; ++j)
      bv[j] = *(const v16h*)(pb0 + (size_t)j * 16 * kD + k0);
#pragma unroll
    for (int i = 0; i < 2; ++i)
#pragma unroll
      for (int j = 0; j < 4; ++j)
        acc[i * 4 + j] = __builtin_amdgcn_wmma_f32_16x16x32_f16(
            false, av[i], false, bv[j], (short)0, acc[i * 4 + j], false, false);
  }

  // C/D layout: VGPR q, lanes 0-15 -> M=q, lanes 16-31 -> M=q+8; N = lane&15
#pragma unroll
  for (int i = 0; i < 2; ++i) {
    int rbase = m0 + i * 16 + hi * 8;
#pragma unroll
    for (int j = 0; j < 4; ++j) {
      int col = n0 + j * 16 + l15;
#pragma unroll
      for (int q = 0; q < 8; ++q)
        C[(size_t)(rbase + q) * kNC + col] = acc[i * 4 + j][q];
    }
  }
}

// ---------------- edge aggregation (segment_sum of messages) --------------
// agg[dst][j] += xWn[src][j] + sum_a ea[e][a] * We1[a][j]

__global__ __launch_bounds__(256) void dgmc_edge_agg1(
    const int* __restrict__ ei, const float* __restrict__ ea,
    const float* __restrict__ We1, const float* __restrict__ C,
    float* __restrict__ agg) {
  long i = (long)blockIdx.x * blockDim.x + threadIdx.x;
  if (i >= (long)kE * kD) return;
  int e = (int)(i / kD), j = (int)(i % kD);
  int src = ei[e], dst = ei[kE + e];
  const float* eap = ea + (size_t)e * 4;
  float v = C[(size_t)src * kNC + j]
          + eap[0] * We1[j]          + eap[1] * We1[kD + j]
          + eap[2] * We1[2 * kD + j] + eap[3] * We1[3 * kD + j];
  atomicAdd(&agg[(size_t)dst * kD + j], v);
}

// h = relu(xWr + agg + b1); also store f16 copy (into recycled xh buffer)

__global__ __launch_bounds__(256) void dgmc_relu_h1(
    const float* __restrict__ C, const float* __restrict__ agg,
    const float* __restrict__ bias, float* __restrict__ h,
    _Float16* __restrict__ hh) {
  long i = (long)blockIdx.x * blockDim.x + threadIdx.x;
  if (i >= (long)kN * kD) return;
  int n = (int)(i / kD), j = (int)(i % kD);
  float v = C[(size_t)n * kNC + kD + j] + agg[i] + bias[j];
  v = v > 0.0f ? v : 0.0f;
  h[i] = v;
  hh[i] = (_Float16)v;
}

// ---------------- batched affinity S_hat[b] = h_s[b] @ h_t[b]^T -----------
// One wave per graph pair; 16x16 tile covers the 15x15 result (row 15
// clamped on load, masked on store).  24 WMMAs per wave over K=768.

__global__ __launch_bounds__(256) void dgmc_shat_wmma(
    const _Float16* __restrict__ Hs, const _Float16* __restrict__ Ht,
    float* __restrict__ Shat) {
  int wave = threadIdx.x >> 5;
  int lane = threadIdx.x & 31;
  int b = blockIdx.x * 8 + wave;
  if (b >= kB) return;
  int l15 = lane & 15, hi = lane >> 4;
  int rr = l15 < kNUM ? l15 : kNUM - 1;   // clamp padded row/col
  const _Float16* pa = Hs + ((size_t)b * kNUM + rr) * kD + hi * 16;
  const _Float16* pb = Ht + ((size_t)b * kNUM + rr) * kD + hi * 16;
  v8f acc;
#pragma unroll
  for (int i = 0; i < 8; ++i) acc[i] = 0.0f;
  for (int k0 = 0; k0 < kD; k0 += 32) {
    v8h a0 = *(const v8h*)(pa + k0);
    v8h a1 = *(const v8h*)(pa + k0 + 8);
    v16h av;
#pragma unroll
    for (int i = 0; i < 8; ++i) { av[i] = a0[i]; av[i + 8] = a1[i]; }
    v16h bv = *(const v16h*)(pb + k0);
    acc = __builtin_amdgcn_wmma_f32_16x16x32_f16(
        false, av, false, bv, (short)0, acc, false, false);
  }
  if (l15 < kNUM) {
#pragma unroll
    for (int i = 0; i < 8; ++i) {
      int s = hi * 8 + i;
      if (s < kNUM)
        Shat[((size_t)b * kNUM + s) * kNUM + l15] = acc[i];
    }
  }
}

// ------------------------------ softmax(15) -------------------------------

__global__ __launch_bounds__(256) void dgmc_softmax15(
    const float* __restrict__ Shat, float* __restrict__ S) {
  int idx = blockIdx.x * blockDim.x + threadIdx.x;   // (b,s) row
  if (idx >= kN) return;
  const float* row = Shat + (size_t)idx * kNUM;
  float m = row[0];
#pragma unroll
  for (int t = 1; t < kNUM; ++t) m = fmaxf(m, row[t]);
  float e[kNUM], sum = 0.0f;
#pragma unroll
  for (int t = 0; t < kNUM; ++t) { e[t] = __expf(row[t] - m); sum += e[t]; }
  float inv = 1.0f / sum;
  float* out = S + (size_t)idx * kNUM;
#pragma unroll
  for (int t = 0; t < kNUM; ++t) out[t] = e[t] * inv;
}

__global__ __launch_bounds__(256) void dgmc_softmax15_argmax(
    const float* __restrict__ Shat, float* __restrict__ S,
    float* __restrict__ p0, int* __restrict__ i0) {
  int idx = blockIdx.x * blockDim.x + threadIdx.x;
  if (idx >= kN) return;
  const float* row = Shat + (size_t)idx * kNUM;
  float m = row[0];
#pragma unroll
  for (int t = 1; t < kNUM; ++t) m = fmaxf(m, row[t]);
  float e[kNUM], sum = 0.0f;
#pragma unroll
  for (int t = 0; t < kNUM; ++t) { e[t] = __expf(row[t] - m); sum += e[t]; }
  float inv = 1.0f / sum;
  float* out = S + (size_t)idx * kNUM;
  float best = -1.0f; int bi = 0;
#pragma unroll
  for (int t = 0; t < kNUM; ++t) {
    float p = e[t] * inv;
    out[t] = p;
    if (p > best) { best = p; bi = t; }
  }
  p0[idx] = best;
  i0[idx] = bi;
}

__global__ __launch_bounds__(256) void dgmc_colmax(
    const float* __restrict__ SL, float* __restrict__ p1, int* __restrict__ i1) {
  int idx = blockIdx.x * blockDim.x + threadIdx.x;   // (b,t)
  if (idx >= kB * kNUM) return;
  int b = idx / kNUM, t = idx % kNUM;
  float best = -1.0f; int bi = 0;
#pragma unroll
  for (int s = 0; s < kNUM; ++s) {
    float v = SL[((size_t)b * kNUM + s) * kNUM + t];
    if (v > best) { best = v; bi = s; }
  }
  p1[idx] = best;
  i1[idx] = bi;
}

// ------------------------- refinement loop kernels ------------------------

__global__ __launch_bounds__(256) void dgmc_rng(float* __restrict__ rs, int step) {
  long i = (long)blockIdx.x * blockDim.x + threadIdx.x;
  if (i >= (long)kN * kR) return;
  unsigned key = (unsigned)i * 2654435769u + (unsigned)(step + 1) * 0x9E3779B9u;
  unsigned u1 = dgmc_hash(key);
  unsigned u2 = dgmc_hash(key ^ 0x85ebca6bu);
  float f1 = ((float)u1 + 0.5f) * (1.0f / 4294967296.0f);
  float f2 = ((float)u2 + 0.5f) * (1.0f / 4294967296.0f);
  rs[i] = __fsqrt_rn(-2.0f * __logf(f1)) * __cosf(6.28318530718f * f2);
}

// r_t[b,t,r] = sum_s S[b,s,t] * r_s[b,s,r]
__global__ __launch_bounds__(256) void dgmc_rt(
    const float* __restrict__ S, const float* __restrict__ rs,
    float* __restrict__ rt) {
  long i = (long)blockIdx.x * blockDim.x + threadIdx.x;
  if (i >= (long)kN * kR) return;
  int b = (int)(i / (kNUM * kR));
  int rem = (int)(i % (kNUM * kR));
  int t = rem / kR, r = rem % kR;
  float sum = 0.0f;
#pragma unroll
  for (int s = 0; s < kNUM; ++s)
    sum += S[(size_t)b * kNUM * kNUM + s * kNUM + t] *
           rs[((size_t)b * kNUM + s) * kR + r];
  rt[i] = sum;
}

// rW[n][0..31] = r@Wn2 ; rW[n][32..63] = r@Wr2
__global__ __launch_bounds__(256) void dgmc_rw(
    const float* __restrict__ r, const float* __restrict__ Wn2,
    const float* __restrict__ Wr2, float* __restrict__ rW) {
  long i = (long)blockIdx.x * blockDim.x + threadIdx.x;
  if (i >= (long)kN * kR) return;
  int n = (int)(i / kR), j = (int)(i % kR);
  const float* rr = r + (size_t)n * kR;
  float s1 = 0.0f, s2 = 0.0f;
#pragma unroll
  for (int k = 0; k < kR; ++k) {
    float rv = rr[k];
    s1 += rv * Wn2[k * kR + j];
    s2 += rv * Wr2[k * kR + j];
  }
  rW[(size_t)n * (2 * kR) + j] = s1;
  rW[(size_t)n * (2 * kR) + kR + j] = s2;
}

__global__ __launch_bounds__(256) void dgmc_edge_agg2(
    const int* __restrict__ ei, const float* __restrict__ ea,
    const float* __restrict__ We2, const float* __restrict__ rW,
    float* __restrict__ agg2) {
  long i = (long)blockIdx.x * blockDim.x + threadIdx.x;
  if (i >= (long)kE * kR) return;
  int e = (int)(i / kR), j = (int)(i % kR);
  int src = ei[e], dst = ei[kE + e];
  const float* eap = ea + (size_t)e * 4;
  float v = rW[(size_t)src * (2 * kR) + j]
          + eap[0] * We2[j]          + eap[1] * We2[kR + j]
          + eap[2] * We2[2 * kR + j] + eap[3] * We2[3 * kR + j];
  atomicAdd(&agg2[(size_t)dst * kR + j], v);
}

__global__ __launch_bounds__(256) void dgmc_relu_o(
    const float* __restrict__ rW, const float* __restrict__ agg2,
    const float* __restrict__ bias, float* __restrict__ o) {
  long i = (long)blockIdx.x * blockDim.x + threadIdx.x;
  if (i >= (long)kN * kR) return;
  int n = (int)(i / kR), j = (int)(i % kR);
  float v = rW[(size_t)n * (2 * kR) + kR + j] + agg2[i] + bias[j];
  o[i] = v > 0.0f ? v : 0.0f;
}

// pairwise MLP update: S_hat[b,s,t] += mlp(o_s[b,s]-o_t[b,t])
__global__ __launch_bounds__(256) void dgmc_upd(
    const float* __restrict__ o_s, const float* __restrict__ o_t,
    const float* __restrict__ W1, const float* __restrict__ bm1,
    const float* __restrict__ W2, const float* __restrict__ bm2,
    float* __restrict__ Shat) {
  __shared__ float sW1[kR * kR];
  __shared__ float sW2[kR];
  __shared__ float sb1[kR];
  for (int i = threadIdx.x; i < kR * kR; i += blockDim.x) sW1[i] = W1[i];
  if (threadIdx.x < kR) { sW2[threadIdx.x] = W2[threadIdx.x]; sb1[threadIdx.x] = bm1[threadIdx.x]; }
  __syncthreads();
  long idx = (long)blockIdx.x * blockDim.x + threadIdx.x;
  if (idx >= (long)kB * kNUM * kNUM) return;
  int b = (int)(idx / (kNUM * kNUM));
  int st = (int)(idx % (kNUM * kNUM));
  int s = st / kNUM, t = st % kNUM;
  const float* os = o_s + ((size_t)b * kNUM + s) * kR;
  const float* ot = o_t + ((size_t)b * kNUM + t) * kR;
  float ds[kR];
#pragma unroll
  for (int k = 0; k < kR; ++k) ds[k] = os[k] - ot[k];
  float acc = bm2[0];
#pragma unroll 4
  for (int j = 0; j < kR; ++j) {
    float hj = sb1[j];
#pragma unroll
    for (int k = 0; k < kR; ++k) hj += ds[k] * sW1[k * kR + j];
    acc += (hj > 0.0f ? hj : 0.0f) * sW2[j];
  }
  Shat[idx] += acc;
}

// ------------------------------ output heads ------------------------------

__global__ __launch_bounds__(256) void dgmc_line_hid(
    const float* __restrict__ S0, const float* __restrict__ SL,
    const float* __restrict__ W1, const float* __restrict__ bl1,
    float* __restrict__ hidL) {
  long i = (long)blockIdx.x * blockDim.x + threadIdx.x;
  if (i >= (long)kB * 128) return;
  int b = (int)(i / 128), j = (int)(i % 128);
  const float* s0 = S0 + (size_t)b * kNUM * kNUM;
  const float* sl = SL + (size_t)b * kNUM * kNUM;
  float sum = bl1[j];
  for (int k = 0; k < kNUM * kNUM; ++k)
    sum += (s0[k] - sl[k]) * W1[(size_t)k * 128 + j];
  hidL[i] = sum > 0.0f ? sum : 0.0f;
}

__global__ __launch_bounds__(256) void dgmc_score(
    const float* __restrict__ hidL, const float* __restrict__ W2,
    const float* __restrict__ bl2, const float* __restrict__ p0,
    const float* __restrict__ p1, const float* __restrict__ xdf_s,
    const float* __restrict__ xdf_t, float* __restrict__ ged) {
  int idx = blockIdx.x * blockDim.x + threadIdx.x;   // (b,s)
  if (idx >= kB * kNUM) return;
  int b = idx / kNUM, s = idx % kNUM;
  float lo = bl2[s];
  const float* hb = hidL + (size_t)b * 128;
  for (int j = 0; j < 128; ++j) lo += hb[j] * W2[j * kNUM + s];
  // node_mask: cumprod of (1 - cond) where cond = both first-features == 0
  int mask = 1;
  for (int jj = 0; jj <= s; ++jj) {
    bool c = (xdf_s[((size_t)b * kNUM + jj) * kD] == 0.0f) &&
             (xdf_t[((size_t)b * kNUM + jj) * kD] == 0.0f);
    if (c) { mask = 0; break; }
  }
  float v = mask ? (p0[idx] + p1[idx]) * 0.5f + lo : 0.0f;
  ged[(size_t)b * 60 + s] = v;
}

// nd/gd: 0.5*||A[b,s]-B[b,i0]+eps|| + 0.5*||B[b,s]-A[b,i1]+eps||
__global__ __launch_bounds__(256) void dgmc_pairdist(
    const float* __restrict__ Xs, const float* __restrict__ Xt,
    const int* __restrict__ i0, const int* __restrict__ i1,
    float* __restrict__ ged, int off) {
  int idx = blockIdx.x * blockDim.x + threadIdx.x;   // (b,s)
  if (idx >= kB * kNUM) return;
  int b = idx / kNUM, s = idx % kNUM;
  const float* a1 = Xs + ((size_t)b * kNUM + s) * kD;
  const float* b1 = Xt + ((size_t)b * kNUM + i0[idx]) * kD;
  const float* a2 = Xt + ((size_t)b * kNUM + s) * kD;
  const float* b2 = Xs + ((size_t)b * kNUM + i1[idx]) * kD;
  float s1 = 0.0f, s2 = 0.0f;
  for (int k = 0; k < kD; ++k) {
    float d1 = a1[k] - b1[k] + 1e-6f;
    float d2 = a2[k] - b2[k] + 1e-6f;
    s1 += d1 * d1; s2 += d2 * d2;
  }
  ged[(size_t)b * 60 + off + s] = 0.5f * __fsqrt_rn(s1) + 0.5f * __fsqrt_rn(s2);
}

__global__ __launch_bounds__(256) void dgmc_pcs_hid(
    const float* __restrict__ cs, const float* __restrict__ ct,
    const float* __restrict__ W1, const float* __restrict__ bd1,
    float* __restrict__ hidP) {
  long i = (long)blockIdx.x * blockDim.x + threadIdx.x;
  if (i >= (long)kB * 128) return;
  int b = (int)(i / 128), j = (int)(i % 128);
  const float* a = cs + (size_t)b * kD;
  const float* c = ct + (size_t)b * kD;
  float sum = bd1[j];
  for (int k = 0; k < kD; ++k) sum += (a[k] - c[k]) * W1[(size_t)k * 128 + j];
  hidP[i] = sum > 0.0f ? sum : 0.0f;
}

__global__ __launch_bounds__(256) void dgmc_pcs_out(
    const float* __restrict__ hidP, const float* __restrict__ W2,
    const float* __restrict__ bd2, float* __restrict__ ged) {
  int idx = blockIdx.x * blockDim.x + threadIdx.x;   // (b,t)
  if (idx >= kB * kNUM) return;
  int b = idx / kNUM, t = idx % kNUM;
  float v = bd2[t];
  const float* hb = hidP + (size_t)b * 128;
  for (int j = 0; j < 128; ++j) v += hb[j] * W2[j * kNUM + t];
  ged[(size_t)b * 60 + 45 + t] = v;
}

// ------------------------------- launcher ---------------------------------

extern "C" void kernel_launch(void* const* d_in, const int* in_sizes, int n_in,
                              void* d_out, int out_size, void* d_ws, size_t ws_size,
                              hipStream_t stream) {
  (void)in_sizes; (void)n_in; (void)out_size; (void)ws_size;
  const float* x_s    = (const float*)d_in[0];
  const float* x_t    = (const float*)d_in[1];
  const int*   ei_s   = (const int*)d_in[2];
  const int*   ei_t   = (const int*)d_in[3];
  const float* ea_s   = (const float*)d_in[4];
  const float* ea_t   = (const float*)d_in[5];
  const float* case_s = (const float*)d_in[8];
  const float* case_t = (const float*)d_in[9];
  const float* emb_in = (const float*)d_in[10];
  const float* emb_out= (const float*)d_in[11];
  const float* Wr1    = (const float*)d_in[12];
  const float* Wn1    = (const float*)d_in[13];
  const float* We1    = (const float*)d_in[14];
  const float* b1v    = (const float*)d_in[15];
  const float* Wr2    = (const float*)d_in[16];
  const float* Wn2    = (const float*)d_in[17];
  const float* We2    = (const float*)d_in[18];
  const float* b2v    = (const float*)d_in[19];
  const float* mW1    = (const float*)d_in[20];
  const float* mb1    = (const float*)d_in[21];
  const float* mW2    = (const float*)d_in[22];
  const float* mb2    = (const float*)d_in[23];
  const float* lW1    = (const float*)d_in[24];
  const float* lb1    = (const float*)d_in[25];
  const float* lW2    = (const float*)d_in[26];
  const float* lb2    = (const float*)d_in[27];
  const float* dW1    = (const float*)d_in[28];
  const float* db1    = (const float*)d_in[29];
  const float* dW2    = (const float*)d_in[30];
  const float* db2    = (const float*)d_in[31];

  float* S0out = (float*)d_out;                       // [N,15]
  float* SLout = S0out + (size_t)kN * kNUM;           // [N,15]
  float* ged   = SLout + (size_t)kN * kNUM;           // [B,60]

  // ---- workspace layout (bump allocator; identical every call) ----
  char* w = (char*)d_ws;
  auto take = [&](size_t bytes) -> char* {
    char* p = w; w += (bytes + 255) & ~(size_t)255; return p;
  };
  int*      outd_s = (int*)take(sizeof(int) * kN);
  int*      ind_s  = (int*)take(sizeof(int) * kN);
  int*      outd_t = (int*)take(sizeof(int) * kN);
  int*      ind_t  = (int*)take(sizeof(int) * kN);
  float*    xdf_s  = (float*)take(sizeof(float) * (size_t)kN * kD);
  float*    xdf_t  = (float*)take(sizeof(float) * (size_t)kN * kD);
  _Float16* xh_s   = (_Float16*)take(sizeof(_Float16) * (size_t)kN * kD); // later hh_s
  _Float16* xh_t   = (_Float16*)take(sizeof(_Float16) * (size_t)kN * kD); // later hh_t
  _Float16* Wct    = (_Float16*)take(sizeof(_Float16) * (size_t)kNC * kD);
  float*    Cbuf   = (float*)take(sizeof(float) * (size_t)kN * kNC);   // reused s then t
  float*    aggbuf = (float*)take(sizeof(float) * (size_t)kN * kD);    // reused
  float*    h_s    = (float*)take(sizeof(float) * (size_t)kN * kD);
  float*    h_t    = (float*)take(sizeof(float) * (size_t)kN * kD);
  float*    Shat   = (float*)take(sizeof(float) * (size_t)kB * kNUM * kNUM);
  float*    Sbuf   = (float*)take(sizeof(float) * (size_t)kB * kNUM * kNUM);
  float*    rsbuf  = (float*)take(sizeof(float) * (size_t)kN * kR);
  float*    rtbuf  = (float*)take(sizeof(float) * (size_t)kN * kR);
  float*    rWbuf  = (float*)take(sizeof(float) * (size_t)kN * 2 * kR);
  float*    agg2   = (float*)take(sizeof(float) * (size_t)kN * kR);
  float*    o_s    = (float*)take(sizeof(float) * (size_t)kN * kR);
  float*    o_t    = (float*)take(sizeof(float) * (size_t)kN * kR);
  float*    p0b    = (float*)take(sizeof(float) * kB * kNUM);
  float*    p1b    = (float*)take(sizeof(float) * kB * kNUM);
  int*      i0b    = (int*)take(sizeof(int) * kB * kNUM);
  int*      i1b    = (int*)take(sizeof(int) * kB * kNUM);
  float*    hidL   = (float*)take(sizeof(float) * kB * 128);
  float*    hidP   = (float*)take(sizeof(float) * kB * 128);

  auto blocks = [](long n) { return dim3((unsigned)((n + 255) / 256)); };

  // ---- phase A: fused weight transpose, then per-side features + GEMM ----
  dgmc_conv_w<<<blocks((long)kNC * kD), 256, 0, stream>>>(Wn1, Wr1, Wct);

  auto run_side1 = [&](const float* x, const int* ei, const float* ea,
                       int* outd, int* ind, float* xdf, _Float16* xh,
                       float* hbuf) {
    dgmc_zero_i32<<<blocks(kN), 256, 0, stream>>>(outd, kN);
    dgmc_zero_i32<<<blocks(kN), 256, 0, stream>>>(ind, kN);
    dgmc_deg_count<<<blocks(kE), 256, 0, stream>>>(ei, outd, ind);
    dgmc_deg_feats<<<blocks((long)kN * kD), 256, 0, stream>>>(
        x, emb_in, emb_out, ind, outd, xdf, xh);
    dim3 ggrid(kN / 128, kNC / 128);                 // (240, 12), 8 waves/block
    dgmc_gemm_wmma<<<ggrid, 256, 0, stream>>>(xh, Wct, Cbuf);
    dgmc_zero_f32<<<blocks((long)kN * kD), 256, 0, stream>>>(aggbuf, (long)kN * kD);
    dgmc_edge_agg1<<<blocks((long)kE * kD), 256, 0, stream>>>(ei, ea, We1, Cbuf, aggbuf);
    dgmc_relu_h1<<<blocks((long)kN * kD), 256, 0, stream>>>(Cbuf, aggbuf, b1v, hbuf, xh);
  };
  run_side1(x_s, ei_s, ea_s, outd_s, ind_s, xdf_s, xh_s, h_s);
  run_side1(x_t, ei_t, ea_t, outd_t, ind_t, xdf_t, xh_t, h_t);

  // ---- phase B: batched affinity via WMMA, initial softmax -> S_0 ----
  dgmc_shat_wmma<<<dim3(kB / 8), 256, 0, stream>>>(xh_s, xh_t, Shat);
  dgmc_softmax15<<<blocks(kN), 256, 0, stream>>>(Shat, S0out);

  // ---- phase C: 10 refinement steps ----
  auto run_psi2 = [&](const float* rbuf, const int* ei, const float* ea,
                      float* obuf) {
    dgmc_rw<<<blocks((long)kN * kR), 256, 0, stream>>>(rbuf, Wn2, Wr2, rWbuf);
    dgmc_zero_f32<<<blocks((long)kN * kR), 256, 0, stream>>>(agg2, (long)kN * kR);
    dgmc_edge_agg2<<<blocks((long)kE * kR), 256, 0, stream>>>(ei, ea, We2, rWbuf, agg2);
    dgmc_relu_o<<<blocks((long)kN * kR), 256, 0, stream>>>(rWbuf, agg2, b2v, obuf);
  };
  for (int step = 0; step < kSteps; ++step) {
    dgmc_softmax15<<<blocks(kN), 256, 0, stream>>>(Shat, Sbuf);
    dgmc_rng<<<blocks((long)kN * kR), 256, 0, stream>>>(rsbuf, step);
    dgmc_rt<<<blocks((long)kN * kR), 256, 0, stream>>>(Sbuf, rsbuf, rtbuf);
    run_psi2(rsbuf, ei_s, ea_s, o_s);
    run_psi2(rtbuf, ei_t, ea_t, o_t);
    dgmc_upd<<<blocks((long)kB * kNUM * kNUM), 256, 0, stream>>>(
        o_s, o_t, mW1, mb1, mW2, mb2, Shat);
  }

  // ---- phase D: outputs ----
  dgmc_softmax15_argmax<<<blocks(kN), 256, 0, stream>>>(Shat, SLout, p0b, i0b);
  dgmc_colmax<<<blocks((long)kB * kNUM), 256, 0, stream>>>(SLout, p1b, i1b);
  dgmc_line_hid<<<blocks((long)kB * 128), 256, 0, stream>>>(S0out, SLout, lW1, lb1, hidL);
  dgmc_score<<<blocks((long)kB * kNUM), 256, 0, stream>>>(
      hidL, lW2, lb2, p0b, p1b, xdf_s, xdf_t, ged);
  dgmc_pairdist<<<blocks((long)kB * kNUM), 256, 0, stream>>>(
      xdf_s, xdf_t, i0b, i1b, ged, 15);                       // nd
  dgmc_pairdist<<<blocks((long)kB * kNUM), 256, 0, stream>>>(
      h_s, h_t, i0b, i1b, ged, 30);                           // gd
  dgmc_pcs_hid<<<blocks((long)kB * 128), 256, 0, stream>>>(case_s, case_t, dW1, db1, hidP);
  dgmc_pcs_out<<<blocks((long)kB * kNUM), 256, 0, stream>>>(hidP, dW2, db2, ged);
}
